// MultiHeadGateAttention_85899345920379
// MI455X (gfx1250) — matrix operand order repaired and checked by reading, not compiled, not run on hardware
//
#include <hip/hip_runtime.h>

// CDNA5 / gfx1250 squared-ReLU causal attention with f16 WMMA.
// B=2, L=2048, H=8, E=D=64.
// Software-pipelined: chunk i+1 staged from global into registers while
// chunk i is computed from LDS; double-buffered LDS, one barrier per chunk.

typedef __attribute__((ext_vector_type(16))) _Float16 v16h;
typedef __attribute__((ext_vector_type(8)))  float    v8f;

#define B_ 2
#define L_ 2048
#define H_ 8
#define E_ 64
#define D_ 64
#define BQ 128      // query rows per workgroup (8 waves x 16 rows)
#define CK 32       // keys per chunk
#define NW 8        // waves per workgroup

// LDS strides in dwords (pairs of f16), chosen odd / conflict-free mod 64
#define KSTRIDE 35  // per key row of K chunk (64 e -> 32 dwords + pad)
#define VSTRIDE 19  // per d row of transposed V chunk (32 keys -> 16 dwords + pad)
#define SSTRIDE 19  // per row of per-wave score scratch (32 keys -> 16 dwords + pad)

#define KBUFSZ (CK * KSTRIDE)
#define VBUFSZ (D_ * VSTRIDE)

union U32H2 { unsigned u; _Float16 h[2]; };

__device__ __forceinline__ unsigned pk2(float a, float b) {
  U32H2 x; x.h[0] = (_Float16)a; x.h[1] = (_Float16)b; return x.u;
}

struct KVregs {
  float4 k0, k1;       // this thread's 8 K elements (one key row, 8 e's)
  float  v0[4], v1[4]; // this thread's 4 V key-pairs (fixed d)
};

// Stage one 32-key chunk from global into registers (coalesced f32 loads).
__device__ __forceinline__ void load_kv(const float* __restrict__ k,
                                        const float* __restrict__ v,
                                        int b, int h, int s0, int tid,
                                        KVregs& r) {
  {
    int key = tid >> 3;       // 0..31
    int seg = tid & 7;        // block of 8 e's
    const float* kb = k + (((size_t)(b * L_ + s0 + key) * H_) + h) * E_ + seg * 8;
    r.k0 = *(const float4*)(kb);
    r.k1 = *(const float4*)(kb + 4);
  }
  {
    int d = tid & 63;
#pragma unroll
    for (int it = 0; it < 4; ++it) {
      int jp = it * 4 + (tid >> 6);   // key pair 0..15
      const float* vb = v + (((size_t)(b * L_ + s0 + 2 * jp) * H_) + h) * D_ + d;
      r.v0[it] = vb[0];
      r.v1[it] = vb[H_ * D_];         // next key, same d
    }
  }
}

// Convert f32 -> packed f16 and store the staged chunk into an LDS buffer.
__device__ __forceinline__ void store_kv(unsigned* __restrict__ kb,
                                         unsigned* __restrict__ vb,
                                         int tid, const KVregs& r) {
  {
    int key = tid >> 3;
    int seg = tid & 7;
    unsigned* dst = &kb[key * KSTRIDE + seg * 4];
    dst[0] = pk2(r.k0.x, r.k0.y);
    dst[1] = pk2(r.k0.z, r.k0.w);
    dst[2] = pk2(r.k1.x, r.k1.y);
    dst[3] = pk2(r.k1.z, r.k1.w);
  }
  {
    int d = tid & 63;
#pragma unroll
    for (int it = 0; it < 4; ++it) {
      int jp = it * 4 + (tid >> 6);
      vb[d * VSTRIDE + jp] = pk2(r.v0[it], r.v1[it]);
    }
  }
}

__global__ __launch_bounds__(256)
void sqrelu_attn_kernel(const float* __restrict__ u,
                        const float* __restrict__ q,
                        const float* __restrict__ k,
                        const float* __restrict__ v,
                        float* __restrict__ out) {
  __shared__ unsigned kbuf[2 * KBUFSZ];          // double-buffered K chunk
  __shared__ unsigned vbuf[2 * VBUFSZ];          // double-buffered V^T chunk
  __shared__ unsigned sbuf[NW * 16 * SSTRIDE];   // per-wave score scratch

  const int tid  = threadIdx.x;
  const int wid  = tid >> 5;
  const int lane = tid & 31;
  const int col  = lane & 15;   // column / row-within-16 selector
  const int hl   = lane >> 4;   // half-wave selector

  const int l0 = blockIdx.x * BQ;
  const int h  = blockIdx.y;
  const int b  = blockIdx.z;

  // ---- load this wave's 16 Q rows straight into WMMA A-operand layout ----
  // A (16x32 f16): lane holds row (lane&15); VGPR v covers
  // e = (lane>>4)*16 + 2v  (+ chunk offset); this kernel's k order, not the ISA table's
  const int qrow = l0 + wid * 16 + col;
  const float* qbase = q + (((size_t)(b * L_ + qrow) * H_) + h) * E_;
  v16h aq0, aq1;
#pragma unroll
  for (int vv = 0; vv < 8; ++vv) {
    int e = hl * 16 + vv * 2;
    float2 f0 = *(const float2*)(qbase + e);        // e-chunk 0..31
    float2 f1 = *(const float2*)(qbase + 32 + e);   // e-chunk 32..63
    aq0[2 * vv] = (_Float16)f0.x; aq0[2 * vv + 1] = (_Float16)f0.y;
    aq1[2 * vv] = (_Float16)f1.x; aq1[2 * vv + 1] = (_Float16)f1.y;
  }

  v8f acc[4] = {};  // 16 rows x 64 d output accumulator (4 C/D tiles)

  const int   qmax  = l0 + wid * 16 + 15;
  const float actsc = 1.0f / (64.0f * 2048.0f);  // scale^2 / S = (1/sqrt(64))^2 / L
  const int   nch   = (l0 + BQ) / CK;

  // ---- prologue: stage + publish chunk 0 ----
  KVregs stage;
  load_kv(k, v, b, h, /*s0=*/0, tid, stage);
  store_kv(&kbuf[0], &vbuf[0], tid, stage);
  __syncthreads();

  for (int ch = 0; ch < nch; ++ch) {
    const int s0  = ch * CK;
    const int sel = ch & 1;
    const bool more = (ch + 1 < nch);

    // Issue global loads for the NEXT chunk now; the s_wait_loadcnt lands
    // after the WMMA compute below (software pipelining).
    if (more) load_kv(k, v, b, h, s0 + CK, tid, stage);

    const unsigned* kcur = &kbuf[sel * KBUFSZ];
    const unsigned* vcur = &vbuf[sel * VBUFSZ];

    if (s0 <= qmax) {   // wave-uniform causal skip (EXEC all-1s inside)
      // ---- GEMM1: 16 q-rows x 32 keys, K-dim = E = 64 (2 chunks) ----
      // B (32x16 f16): lane holds key column (lane&15); VGPR v covers
      // e = (lane>>4)*16 + 2v (+ e-chunk offset); pairs contiguous in kbuf; k order chosen to match A.
      v8f s_t0 = {}, s_t1 = {};
#pragma unroll
      for (int c = 0; c < 2; ++c) {
        v16h bk0, bk1;
#pragma unroll
        for (int vv = 0; vv < 8; ++vv) {
          int j = c * 16 + hl * 8 + vv;
          U32H2 x0; x0.u = kcur[col * KSTRIDE + j];         // keys 0..15
          U32H2 x1; x1.u = kcur[(16 + col) * KSTRIDE + j];  // keys 16..31
          bk0[2 * vv] = x0.h[0]; bk0[2 * vv + 1] = x0.h[1];
          bk1[2 * vv] = x1.h[0]; bk1[2 * vv + 1] = x1.h[1];
        }
        const v16h a = (c == 0) ? aq0 : aq1;
        s_t0 = __builtin_amdgcn_wmma_f32_16x16x32_f16(false, a, false, bk0,
                                                      (short)0, s_t0, false, false);
        s_t1 = __builtin_amdgcn_wmma_f32_16x16x32_f16(false, a, false, bk1,
                                                      (short)0, s_t1, false, false);
      }

      // ---- activation: relu(x)^2 * scale^2 / S, causal mask; stash to LDS ----
      // C/D layout: element (m,n): m = r + 8*(lane>>4), n = lane&15.
      _Float16* sh = (_Float16*)&sbuf[wid * 16 * SSTRIDE];
#pragma unroll
      for (int r = 0; r < 8; ++r) {
        int   m   = r + 8 * hl;
        int   qr  = l0 + wid * 16 + m;
        int   k0g = s0 + col;
        int   k1g = k0g + 16;
        float x0 = s_t0[r]; x0 = x0 > 0.f ? x0 : 0.f; x0 = (k0g <= qr) ? x0 * x0 * actsc : 0.f;
        float x1 = s_t1[r]; x1 = x1 > 0.f ? x1 : 0.f; x1 = (k1g <= qr) ? x1 * x1 * actsc : 0.f;
        sh[m * (2 * SSTRIDE) + col]      = (_Float16)x0;
        sh[m * (2 * SSTRIDE) + 16 + col] = (_Float16)x1;
      }
      // LDS DS ops are in-order within a wave; just stop compiler reordering.
      asm volatile("" ::: "memory");

      // ---- rebuild scores in A-operand layout (16x32 f16) ----
      v16h as;
#pragma unroll
      for (int vv = 0; vv < 8; ++vv) {
        int j = hl * 8 + vv;   // = K/2 for this VGPR
        U32H2 x; x.u = sbuf[wid * 16 * SSTRIDE + col * SSTRIDE + j];
        as[2 * vv] = x.h[0]; as[2 * vv + 1] = x.h[1];
      }

      // ---- GEMM2: acc += A(16x32) x V(32x64), per 16-wide d tile ----
      // B: lane holds d column; VGPR v covers keys (lane>>4)*16 + 2v (+1), k order matching A,
      // contiguous in transposed vbuf.
#pragma unroll
      for (int t = 0; t < 4; ++t) {
        v16h bv;
#pragma unroll
        for (int vv = 0; vv < 8; ++vv) {
          int d = t * 16 + col;
          U32H2 x; x.u = vcur[d * VSTRIDE + hl * 8 + vv];
          bv[2 * vv] = x.h[0]; bv[2 * vv + 1] = x.h[1];
        }
        acc[t] = __builtin_amdgcn_wmma_f32_16x16x32_f16(false, as, false, bv,
                                                        (short)0, acc[t], false, false);
      }
    }

    // Publish next chunk into the buffer nobody reads this iteration.
    if (more) store_kv(&kbuf[(sel ^ 1) * KBUFSZ], &vbuf[(sel ^ 1) * VBUFSZ],
                       tid, stage);
    // Single barrier: publishes buf[sel^1] for iteration ch+1 AND guarantees
    // all waves finished reading buf[sel] before it is overwritten at ch+2.
    __syncthreads();
  }

  // ---- epilogue: out = u * (A @ V), coalesced over lanes ----
#pragma unroll
  for (int t = 0; t < 4; ++t) {
#pragma unroll
    for (int r = 0; r < 8; ++r) {
      int row = l0 + wid * 16 + r + 8 * hl;
      int d   = t * 16 + col;
      size_t idx = (((size_t)(b * L_ + row) * H_) + h) * D_ + d;
      out[idx] = u[idx] * acc[t][r];
    }
  }
}

extern "C" void kernel_launch(void* const* d_in, const int* in_sizes, int n_in,
                              void* d_out, int out_size, void* d_ws, size_t ws_size,
                              hipStream_t stream) {
  const float* u = (const float*)d_in[0];
  const float* q = (const float*)d_in[1];
  const float* k = (const float*)d_in[2];
  const float* v = (const float*)d_in[3];
  // d_in[4] = attn_mask (causal, strict upper triangle) -> computed analytically
  float* out = (float*)d_out;

  dim3 grid(L_ / BQ, H_, B_);
  sqrelu_attn_kernel<<<grid, 256, 0, stream>>>(u, q, k, v, out);
}
